// DiTBlock_53214644797675
// MI455X (gfx1250) — matrix operand order repaired and checked by reading, not compiled, not run on hardware
//
#include <hip/hip_runtime.h>
#include <hip/hip_bf16.h>
#include <math.h>

#define DIM      1024
#define HEADS    16
#define HEAD_DIM 64
#define MLP_DIM  4096
#define BATCH    2
#define SEQ      2048
#define ROWS     (BATCH * SEQ)

typedef __attribute__((ext_vector_type(16))) _Float16 v16h;
typedef __attribute__((ext_vector_type(8)))  _Float16 v8h;
typedef __attribute__((ext_vector_type(8)))  float    v8f;

union Frag16 { v16h v; v8h h[2]; };

__device__ __forceinline__ v8f wmma16(v16h a, v16h b, v8f c) {
  // D = A(16x32 f16) x B(32x16 f16) + C(16x16 f32)
  return __builtin_amdgcn_wmma_f32_16x16x32_f16(false, a, false, b, (short)0, c,
                                                false, false);
}

// A-matrix fragment (16x32 f16, row-major source, leading dim = lda).
// ISA layout: lanes 0-15 hold row m, K = k..k+7 then k+16..k+23;
//             lanes 16-31 hold row m, K = k+8..k+15 then k+24..k+31.
__device__ __forceinline__ v16h load_a_frag(const _Float16* A, int lda, int m0, int k0) {
  const int lane = threadIdx.x & 31;
  const int m = m0 + (lane & 15);
  const int k = k0 + ((lane >> 4) << 3);
  const _Float16* p = A + (size_t)m * lda + k;
  Frag16 f;
  f.h[0] = *(const v8h*)(p);
  f.h[1] = *(const v8h*)(p + 16);
  return f.v;
}

// B-matrix fragment (32x16 f16) from "WT" storage: column n of B stored
// contiguously along K (i.e. WT[n][k], leading dim = ldb).
// ISA layout: lanes 0-15 hold column n, K = k..k+7 then k+16..k+23;
//             lanes 16-31 hold column n, K = k+8..k+15 then k+24..k+31.
__device__ __forceinline__ v16h load_b_frag(const _Float16* WT, int ldb, int n0, int k0) {
  const int lane = threadIdx.x & 31;
  const int n = n0 + (lane & 15);
  const int k = k0 + ((lane >> 4) << 3);
  const _Float16* p = WT + (size_t)n * ldb + k;
  Frag16 f;
  f.h[0] = *(const v8h*)(p);
  f.h[1] = *(const v8h*)(p + 16);
  return f.v;
}

// ---------------------------------------------------------------------------
// Weight transpose + cast: W[K][Nc] f32  ->  WT[Nc][K] f16
// ---------------------------------------------------------------------------
__global__ void transpose_f32_to_f16(const float* __restrict__ W,
                                     _Float16* __restrict__ WT, int K, int Nc) {
  __shared__ float tile[16][17];
  const int n0 = blockIdx.x * 16, k0 = blockIdx.y * 16;
  tile[threadIdx.y][threadIdx.x] =
      W[(size_t)(k0 + threadIdx.y) * Nc + n0 + threadIdx.x];
  __syncthreads();
  WT[(size_t)(n0 + threadIdx.y) * K + k0 + threadIdx.x] =
      (_Float16)tile[threadIdx.x][threadIdx.y];
}

// ---------------------------------------------------------------------------
// RMSNorm (F.normalize * sqrt(D) * g) -> f16 activations. One block per row.
// ---------------------------------------------------------------------------
__global__ void __launch_bounds__(256)
rmsnorm_to_f16(const float* __restrict__ x, const float* __restrict__ g,
               _Float16* __restrict__ out) {
  const int row = blockIdx.x;
  const float* xr = x + (size_t)row * DIM;
  float v[4];
  float ss = 0.f;
#pragma unroll
  for (int i = 0; i < 4; ++i) {
    v[i] = xr[threadIdx.x + 256 * i];
    ss += v[i] * v[i];
  }
#pragma unroll
  for (int off = 16; off > 0; off >>= 1) ss += __shfl_xor(ss, off, 32);
  __shared__ float red[8];
  if ((threadIdx.x & 31) == 0) red[threadIdx.x >> 5] = ss;
  __syncthreads();
  float tot = 0.f;
#pragma unroll
  for (int w = 0; w < 8; ++w) tot += red[w];
  const float scale = 32.0f * g[0] / fmaxf(sqrtf(tot), 1e-12f);  // sqrt(1024)=32
#pragma unroll
  for (int i = 0; i < 4; ++i)
    out[(size_t)row * DIM + threadIdx.x + 256 * i] = (_Float16)(v[i] * scale);
}

// ---------------------------------------------------------------------------
// q/k head-norm + scale folding; K laid row-major, V transposed per (b,h)
// so all attention B-fragments are contiguous per-lane loads.
//   qhat = q/||q|| * g_q          (the 1/sqrt(d) and sqrt(d) factors cancel)
//   khat = k/||k|| * 8 * g_k
//   vT[b,h][d][n] = v
// One wave per (row, head): blockDim = (32,16), grid = ROWS.
// ---------------------------------------------------------------------------
__global__ void qk_prep(const float* __restrict__ qkv, const float* __restrict__ gq,
                        const float* __restrict__ gk, _Float16* __restrict__ qhat,
                        _Float16* __restrict__ khat, _Float16* __restrict__ vT) {
  const int row = blockIdx.x;        // b*SEQ + n
  const int h = threadIdx.y;
  const int lane = threadIdx.x;
  const int b = row / SEQ, n = row % SEQ;
  const float* base = qkv + (size_t)row * (3 * DIM) + h * HEAD_DIM;
  const int d0 = lane * 2;
  const float q0 = base[d0],            q1 = base[d0 + 1];
  const float k0 = base[DIM + d0],      k1 = base[DIM + d0 + 1];
  const float v0 = base[2 * DIM + d0],  v1 = base[2 * DIM + d0 + 1];
  float sq = q0 * q0 + q1 * q1;
  float sk = k0 * k0 + k1 * k1;
#pragma unroll
  for (int off = 16; off > 0; off >>= 1) {
    sq += __shfl_xor(sq, off, 32);
    sk += __shfl_xor(sk, off, 32);
  }
  const float qs = gq[0] / fmaxf(sqrtf(sq), 1e-12f);
  const float ks = 8.0f * gk[0] / fmaxf(sqrtf(sk), 1e-12f);  // sqrt(64)=8
  const size_t bh = (size_t)(b * HEADS + h);
  _Float16* qo = qhat + (bh * SEQ + n) * HEAD_DIM + d0;
  _Float16* ko = khat + (bh * SEQ + n) * HEAD_DIM + d0;
  qo[0] = (_Float16)(q0 * qs); qo[1] = (_Float16)(q1 * qs);
  ko[0] = (_Float16)(k0 * ks); ko[1] = (_Float16)(k1 * ks);
  _Float16* vo = vT + bh * HEAD_DIM * SEQ;  // [64][SEQ]
  vo[(size_t)d0 * SEQ + n]       = (_Float16)v0;
  vo[(size_t)(d0 + 1) * SEQ + n] = (_Float16)v1;
}

// ---------------------------------------------------------------------------
// Flash attention with BOUNDED-SHIFT softmax (no cross-lane reductions).
// |score| <= ||qhat||*||khat|| = 8*|g_q*g_k| =: M, and softmax is shift-
// invariant, so we use exp(s - M) with the fixed bound instead of an online
// row max. Row sums are produced by one extra WMMA against a ones-matrix,
// which leaves the sum replicated across all 16 columns of the C tile, i.e.
// already in the right lanes. One block per (64 q rows, b*H); 4 waves, each
// owning a 16-row Q stripe; P restaged through per-wave LDS into A-fragment
// order for the P@V WMMAs.
// ---------------------------------------------------------------------------
__global__ void __launch_bounds__(128)
flash_attn(const _Float16* __restrict__ qhat, const _Float16* __restrict__ khat,
           const _Float16* __restrict__ vT, const float* __restrict__ gq,
           const float* __restrict__ gk, _Float16* __restrict__ hO) {
  const int lane = threadIdx.x;
  const int wave = threadIdx.y;
  const int bh = blockIdx.y;
  const int b = bh / HEADS, h = bh % HEADS;
  const int q0 = blockIdx.x * 64 + wave * 16;  // seq-local base of this wave
  const _Float16* Q = qhat + (size_t)bh * SEQ * HEAD_DIM;
  const _Float16* K = khat + (size_t)bh * SEQ * HEAD_DIM;
  const _Float16* V = vT + (size_t)bh * HEAD_DIM * SEQ;

  const float M = 8.0f * fabsf(gq[0]) * fabsf(gk[0]);  // score upper bound

  const v16h aQ0 = load_a_frag(Q, HEAD_DIM, q0, 0);
  const v16h aQ1 = load_a_frag(Q, HEAD_DIM, q0, 32);

  v16h ones;
#pragma unroll
  for (int i = 0; i < 16; ++i) ones[i] = (_Float16)1.0f;

  v8f acc[4] = {};  // O tile: 16 q rows x 64 dims (4 x 16-col tiles)
  v8f accs = {};    // row sums, replicated over columns

  __shared__ _Float16 plds[4][16][72];  // per-wave P tile, padded stride
  _Float16* pw = &plds[wave][0][0];
  const int col = lane & 15, hi = lane >> 4;

  for (int kb = 0; kb < SEQ; kb += 64) {
    // ---- scores S = qhat . khat for a 16x64 tile (8 WMMAs) ----
    v8f s[4];
#pragma unroll
    for (int t = 0; t < 4; ++t) {
      const v16h b0 = load_b_frag(K, HEAD_DIM, kb + 16 * t, 0);
      const v16h b1 = load_b_frag(K, HEAD_DIM, kb + 16 * t, 32);
      v8f z = {};
      s[t] = wmma16(aQ0, b0, z);
      s[t] = wmma16(aQ1, b1, s[t]);
    }
    // ---- P = exp(S - M), purely element-wise ----
#pragma unroll
    for (int t = 0; t < 4; ++t)
#pragma unroll
      for (int r = 0; r < 8; ++r) s[t][r] = __expf(s[t][r] - M);
    // ---- restage P (C-layout regs) into LDS as row-major f16 ----
    __syncthreads();
#pragma unroll
    for (int t = 0; t < 4; ++t)
#pragma unroll
      for (int r = 0; r < 8; ++r)
        pw[(size_t)(r + 8 * hi) * 72 + 16 * t + col] = (_Float16)s[t][r];
    __syncthreads();
    // ---- O += P @ V ; rowsum += P @ 1  (10 WMMAs) ----
#pragma unroll
    for (int st = 0; st < 2; ++st) {
      const v16h aP = load_a_frag(pw, 72, 0, 32 * st);
      accs = wmma16(aP, ones, accs);
#pragma unroll
      for (int dt = 0; dt < 4; ++dt) {
        const v16h bV = load_b_frag(V, SEQ, 16 * dt, kb + 32 * st);
        acc[dt] = wmma16(aP, bV, acc[dt]);
      }
    }
  }
  // ---- normalize and scatter into [B,N,D] f16 for the proj GEMM ----
  float inv[8];
#pragma unroll
  for (int r = 0; r < 8; ++r) inv[r] = 1.0f / fmaxf(accs[r], 1e-37f);
#pragma unroll
  for (int dt = 0; dt < 4; ++dt)
#pragma unroll
    for (int r = 0; r < 8; ++r) {
      const int m = q0 + r + 8 * hi;
      const float o = acc[dt][r] * inv[r];
      hO[(size_t)(b * SEQ + m) * DIM + h * HEAD_DIM + 16 * dt + col] = (_Float16)o;
    }
}

// ---------------------------------------------------------------------------
// f16 WMMA GEMM: out = epilogue(A[M,K] @ WT^T + bias).
// Block = 4 waves (2x2), block tile 64(M) x 256(N); wave tile 32x128:
// 16 WMMAs per 10 fragment loads per K-step.
// ---------------------------------------------------------------------------
template <bool RESID, bool GELU_T, bool OUT32, bool OUT16>
__global__ void __launch_bounds__(128)
gemm_wmma(const _Float16* __restrict__ A, const _Float16* __restrict__ WT,
          const float* __restrict__ bias, const float* __restrict__ resid,
          float* __restrict__ out32, _Float16* __restrict__ out16,
          int M, int Nc, int K) {
  const int lane = threadIdx.x;
  const int wave = threadIdx.y;
  const int m0 = blockIdx.y * 64 + (wave >> 1) * 32;
  const int n0 = blockIdx.x * 256 + (wave & 1) * 128;
  v8f acc[2][8] = {};
  for (int k0 = 0; k0 < K; k0 += 32) {
    if (k0 + 32 < K) {  // global_prefetch_b8 of next K-slice
      __builtin_prefetch(A + (size_t)(m0 + (lane & 15)) * K + k0 + 32, 0, 1);
      __builtin_prefetch(WT + (size_t)(n0 + (lane & 15)) * K + k0 + 32, 0, 1);
    }
    const v16h a0 = load_a_frag(A, K, m0, k0);
    const v16h a1 = load_a_frag(A, K, m0 + 16, k0);
#pragma unroll
    for (int t = 0; t < 8; ++t) {
      const v16h bf = load_b_frag(WT, K, n0 + 16 * t, k0);
      acc[0][t] = wmma16(a0, bf, acc[0][t]);
      acc[1][t] = wmma16(a1, bf, acc[1][t]);
    }
  }
  const int col = lane & 15, hi = lane >> 4;
#pragma unroll
  for (int i = 0; i < 2; ++i)
#pragma unroll
    for (int t = 0; t < 8; ++t) {
      const int n = n0 + 16 * t + col;
      const float bv = bias[n];
#pragma unroll
      for (int r = 0; r < 8; ++r) {
        const int m = m0 + 16 * i + r + 8 * hi;
        float v = acc[i][t][r] + bv;
        if (RESID) v += resid[(size_t)m * Nc + n];
        if (GELU_T) {
          const float u = 0.7978845608028654f * (v + 0.044715f * v * v * v);
          v = 0.5f * v * (1.0f + tanhf(u));
        }
        if (OUT32) out32[(size_t)m * Nc + n] = v;
        if (OUT16) out16[(size_t)m * Nc + n] = (_Float16)v;
      }
    }
}

// ---------------------------------------------------------------------------
extern "C" void kernel_launch(void* const* d_in, const int* in_sizes, int n_in,
                              void* d_out, int out_size, void* d_ws, size_t ws_size,
                              hipStream_t stream) {
  (void)in_sizes; (void)n_in; (void)out_size; (void)ws_size;
  const float* x      = (const float*)d_in[0];
  const float* w_qkv  = (const float*)d_in[1];
  const float* b_qkv  = (const float*)d_in[2];
  const float* w_proj = (const float*)d_in[3];
  const float* b_proj = (const float*)d_in[4];
  const float* g1     = (const float*)d_in[5];
  const float* g2     = (const float*)d_in[6];
  const float* gq     = (const float*)d_in[7];
  const float* gk     = (const float*)d_in[8];
  const float* w_fc1  = (const float*)d_in[9];
  const float* b_fc1  = (const float*)d_in[10];
  const float* w_fc2  = (const float*)d_in[11];
  const float* b_fc2  = (const float*)d_in[12];
  float* out = (float*)d_out;

  char* ws = (char*)d_ws;
  size_t off = 0;
  auto alloc = [&](size_t bytes) -> void* {
    void* p = ws + off;
    off += (bytes + 255) & ~(size_t)255;
    return p;
  };
  _Float16* hA     = (_Float16*)alloc((size_t)ROWS * DIM * 2);
  _Float16* wqkvT  = (_Float16*)alloc((size_t)(3 * DIM) * DIM * 2);
  _Float16* wprojT = (_Float16*)alloc((size_t)DIM * DIM * 2);
  _Float16* wfc1T  = (_Float16*)alloc((size_t)MLP_DIM * DIM * 2);
  _Float16* wfc2T  = (_Float16*)alloc((size_t)DIM * MLP_DIM * 2);
  float*    qkvb   = (float*)alloc((size_t)ROWS * 3 * DIM * 4);
  _Float16* qhat   = (_Float16*)alloc((size_t)BATCH * HEADS * SEQ * HEAD_DIM * 2);
  _Float16* khat   = (_Float16*)alloc((size_t)BATCH * HEADS * SEQ * HEAD_DIM * 2);
  _Float16* vT     = (_Float16*)alloc((size_t)BATCH * HEADS * SEQ * HEAD_DIM * 2);
  _Float16* hO     = (_Float16*)alloc((size_t)ROWS * DIM * 2);
  float*    x1     = (float*)alloc((size_t)ROWS * DIM * 4);
  _Float16* hM     = (_Float16*)alloc((size_t)ROWS * MLP_DIM * 2);

  const dim3 tb16(16, 16);
  // Weight casts/transposes: W[K][Nc] f32 -> WT[Nc][K] f16
  transpose_f32_to_f16<<<dim3((3 * DIM) / 16, DIM / 16), tb16, 0, stream>>>(w_qkv, wqkvT, DIM, 3 * DIM);
  transpose_f32_to_f16<<<dim3(DIM / 16, DIM / 16), tb16, 0, stream>>>(w_proj, wprojT, DIM, DIM);
  transpose_f32_to_f16<<<dim3(MLP_DIM / 16, DIM / 16), tb16, 0, stream>>>(w_fc1, wfc1T, DIM, MLP_DIM);
  transpose_f32_to_f16<<<dim3(DIM / 16, MLP_DIM / 16), tb16, 0, stream>>>(w_fc2, wfc2T, MLP_DIM, DIM);

  // Attention branch
  rmsnorm_to_f16<<<ROWS, 256, 0, stream>>>(x, g1, hA);
  gemm_wmma<false, false, true, false>
      <<<dim3((3 * DIM) / 256, ROWS / 64), dim3(32, 4), 0, stream>>>(
          hA, wqkvT, b_qkv, nullptr, qkvb, nullptr, ROWS, 3 * DIM, DIM);
  qk_prep<<<ROWS, dim3(32, 16), 0, stream>>>(qkvb, gq, gk, qhat, khat, vT);
  flash_attn<<<dim3(SEQ / 64, BATCH * HEADS), dim3(32, 4), 0, stream>>>(
      qhat, khat, vT, gq, gk, hO);
  gemm_wmma<true, false, true, false>
      <<<dim3(DIM / 256, ROWS / 64), dim3(32, 4), 0, stream>>>(
          hO, wprojT, b_proj, x, x1, nullptr, ROWS, DIM, DIM);

  // MLP branch
  rmsnorm_to_f16<<<ROWS, 256, 0, stream>>>(x1, g2, hA);
  gemm_wmma<false, true, false, true>
      <<<dim3(MLP_DIM / 256, ROWS / 64), dim3(32, 4), 0, stream>>>(
          hA, wfc1T, b_fc1, nullptr, nullptr, hM, ROWS, MLP_DIM, DIM);
  gemm_wmma<true, false, true, false>
      <<<dim3(DIM / 256, ROWS / 64), dim3(32, 4), 0, stream>>>(
          hM, wfc2T, b_fc2, x1, out, nullptr, ROWS, DIM, MLP_DIM);
}
